// CosAttention_10582799417399
// MI455X (gfx1250) — matrix operand order repaired and checked by reading, not compiled, not run on hardware
//
#include <hip/hip_runtime.h>

typedef __attribute__((ext_vector_type(16))) _Float16 v16h;
typedef __attribute__((ext_vector_type(8)))  float    v8f;
typedef __attribute__((ext_vector_type(2)))  _Float16 h2;

// Problem dims (fixed by reference): B=4 H=12 N=8192 D=64
constexpr int Bc = 4, Hc = 12, Nc = 8192, Dc = 64;
constexpr int BH = Bc * Hc;          // 48
constexpr int CH = 16;               // split-K chunks over N
constexpr int CHROWS = Nc / CH;      // 512 rows per chunk
constexpr float SCALE = 0.105112051906714f;  // 8192^(-1/4)

constexpr int KSTR = 34;   // LDS row stride (halves), 17-bank stride -> conflict free
constexpr int QSTR = 66;   // phase-2 Q row stride (halves), 33-bank stride
constexpr int TSTR = 66;   // phase-2 KtV^T row stride (halves)

// ---------------------------------------------------------------------------
// Phase 1: partial KtV[d][e] = sum_n Kf[n][d] * Vm[n][e]   (per bh, per chunk)
// ---------------------------------------------------------------------------
__global__ __launch_bounds__(256) void cosattn_p1(
    const float* __restrict__ K, const float* __restrict__ V,
    const float* __restrict__ mask, float* __restrict__ part) {
  __shared__ _Float16 sK[64 * KSTR];  // Kf^T : [d][n_local]
  __shared__ _Float16 sV[64 * KSTR];  // Vm^T : [e][n_local]

  const int blk = blockIdx.x;
  const int bh = blk / CH, ch = blk % CH;
  const int bb = bh / Hc;
  const int nbase = ch * CHROWS;

  const int t = threadIdx.x;
  const int lane = t & 31;
  const int wave = t >> 5;
  const int r  = t >> 3;          // row 0..31 within 32-row tile
  const int cb = (t & 7) * 8;     // col base (8 floats per thread)

  // wave -> (d-block, e-half) of the 64x64 output
  const int d0  = (wave >> 1) * 16;
  const int e0  = (wave & 1) * 32;
  const int m   = lane & 15;
  const int hi  = lane >> 4;

  v8f c0 = {}; v8f c1 = {};

  for (int tile = 0; tile < CHROWS / 32; ++tile) {
    const int n = nbase + tile * 32 + r;
    const float4* kp4 = (const float4*)(K + ((size_t)bh * Nc + n) * Dc + cb);
    const float4* vp4 = (const float4*)(V + ((size_t)bh * Nc + n) * Dc + cb);
    float4 ka = kp4[0], kb4 = kp4[1];
    float4 va = vp4[0], vb4 = vp4[1];
    const float mv = mask[bb * Nc + n];

    float kv[8] = {ka.x, ka.y, ka.z, ka.w, kb4.x, kb4.y, kb4.z, kb4.w};
    float vv[8] = {va.x, va.y, va.z, va.w, vb4.x, vb4.y, vb4.z, vb4.w};

    float ss = 0.f;
#pragma unroll
    for (int i = 0; i < 8; ++i) ss += kv[i] * kv[i];
    ss += __shfl_xor(ss, 1, 32);
    ss += __shfl_xor(ss, 2, 32);
    ss += __shfl_xor(ss, 4, 32);
    const float fk = SCALE / fmaxf(sqrtf(ss), 1e-12f) * mv;

    __syncthreads();  // previous iteration's fragment reads done
#pragma unroll
    for (int i = 0; i < 8; ++i) {
      sK[(cb + i) * KSTR + r] = (_Float16)(kv[i] * fk);  // transposed store
      sV[(cb + i) * KSTR + r] = (_Float16)(vv[i] * mv);
    }
    __syncthreads();

    // A fragment: A[m][k] = Kf[n0+k][d0+m]  (16-bit A 16x32 layout)
    v16h a;
#pragma unroll
    for (int j = 0; j < 8; ++j) {
      const int kb = ((j & 4) << 2) + (hi << 3) + ((j & 3) << 1);
      h2 p = *(const h2*)&sK[(d0 + m) * KSTR + kb];
      a[2 * j] = p.x; a[2 * j + 1] = p.y;
    }
    // B fragments: B[k][n] = Vm[n0+k][e0+n]  (16-bit B 32x16 layout)
    v16h b0, b1;
#pragma unroll
    for (int j = 0; j < 8; ++j) {
      const int kb = ((j & 4) << 2) + (hi << 3) + ((j & 3) << 1);
      h2 p = *(const h2*)&sV[(e0 + m) * KSTR + kb];
      b0[2 * j] = p.x; b0[2 * j + 1] = p.y;
      h2 q = *(const h2*)&sV[(e0 + 16 + m) * KSTR + kb];
      b1[2 * j] = q.x; b1[2 * j + 1] = q.y;
    }
    c0 = __builtin_amdgcn_wmma_f32_16x16x32_f16(false, a, false, b0, (short)0, c0, false, false);
    c1 = __builtin_amdgcn_wmma_f32_16x16x32_f16(false, a, false, b1, (short)0, c1, false, false);
  }

  // store partial 64x64 (C layout: VGPR r -> M=r / r+8)
  float* pp = part + (size_t)blk * 64 * 64;
#pragma unroll
  for (int rr = 0; rr < 8; ++rr) {
    const int d = d0 + rr + hi * 8;
    pp[d * 64 + e0 + m]      = c0[rr];
    pp[d * 64 + e0 + 16 + m] = c1[rr];
  }
}

// ---------------------------------------------------------------------------
// Phase 1b: deterministic reduction of CH partials -> KtV (f32)
// ---------------------------------------------------------------------------
__global__ __launch_bounds__(256) void cosattn_p1r(
    const float* __restrict__ part, float* __restrict__ ktv) {
  const int bh = blockIdx.x;
  for (int i = threadIdx.x; i < 4096; i += 256) {
    float s = 0.f;
#pragma unroll
    for (int c = 0; c < CH; ++c)
      s += part[((size_t)(bh * CH + c)) * 4096 + i];
    ktv[(size_t)bh * 4096 + i] = s;
  }
}

// ---------------------------------------------------------------------------
// Phase 2: out[n][e] = sum_d Qf[n][d] * KtV[d][e]
// ---------------------------------------------------------------------------
__global__ __launch_bounds__(256) void cosattn_p2(
    const float* __restrict__ Q, const float* __restrict__ ktv,
    float* __restrict__ out) {
  __shared__ _Float16 sT[64 * TSTR];    // KtV^T : [e][d]
  __shared__ _Float16 sQ[128 * QSTR];   // Qf    : [n_local][d]

  const int bh = blockIdx.y;
  const int nbase = blockIdx.x * 128;
  const int t = threadIdx.x;

  // stage KtV (f32 -> f16, transposed)
  const float* kp = ktv + (size_t)bh * 4096;
  for (int i = t; i < 4096; i += 256) {
    const int d = i >> 6, e = i & 63;
    sT[e * TSTR + d] = (_Float16)kp[i];
  }

  // stage + normalize 128 Q rows (2 threads per row, 32 floats each)
  {
    const int r  = t >> 1;
    const int cb = (t & 1) * 32;
    const int n  = nbase + r;
    const float4* qp4 = (const float4*)(Q + ((size_t)bh * Nc + n) * Dc + cb);
    float qv[32];
#pragma unroll
    for (int i = 0; i < 8; ++i) {
      float4 v = qp4[i];
      qv[4 * i] = v.x; qv[4 * i + 1] = v.y; qv[4 * i + 2] = v.z; qv[4 * i + 3] = v.w;
    }
    float ss = 0.f;
#pragma unroll
    for (int i = 0; i < 32; ++i) ss += qv[i] * qv[i];
    ss += __shfl_xor(ss, 1, 32);   // partner thread of same row
    const float fq = SCALE / fmaxf(sqrtf(ss), 1e-12f);
#pragma unroll
    for (int i = 0; i < 32; i += 2) {
      h2 p = {(_Float16)(qv[i] * fq), (_Float16)(qv[i + 1] * fq)};
      *(h2*)&sQ[r * QSTR + cb + i] = p;
    }
  }
  __syncthreads();

  const int lane = t & 31, wave = t >> 5;
  const int m = lane & 15, hi = lane >> 4;
  const int rowb = wave * 16;   // each wave: 16 rows x 64 cols

  v8f c[4] = {{}, {}, {}, {}};
#pragma unroll
  for (int ks = 0; ks < 2; ++ks) {         // K = 64 in two K=32 steps
    v16h a;
#pragma unroll
    for (int j = 0; j < 8; ++j) {
      const int kb = ks * 32 + ((j & 4) << 2) + (hi << 3) + ((j & 3) << 1);
      h2 p = *(const h2*)&sQ[(rowb + m) * QSTR + kb];
      a[2 * j] = p.x; a[2 * j + 1] = p.y;
    }
#pragma unroll
    for (int e = 0; e < 4; ++e) {
      v16h bf;
#pragma unroll
      for (int j = 0; j < 8; ++j) {
        const int kb = ks * 32 + ((j & 4) << 2) + (hi << 3) + ((j & 3) << 1);
        h2 p = *(const h2*)&sT[(e * 16 + m) * TSTR + kb];
        bf[2 * j] = p.x; bf[2 * j + 1] = p.y;
      }
      c[e] = __builtin_amdgcn_wmma_f32_16x16x32_f16(false, a, false, bf, (short)0, c[e], false, false);
    }
  }

  float* op = out + ((size_t)bh * Nc + nbase) * Dc;
#pragma unroll
  for (int e = 0; e < 4; ++e)
#pragma unroll
    for (int rr = 0; rr < 8; ++rr) {
      const int row = rowb + rr + hi * 8;
      op[row * 64 + e * 16 + m] = c[e][rr];
    }
}

// ---------------------------------------------------------------------------
extern "C" void kernel_launch(void* const* d_in, const int* in_sizes, int n_in,
                              void* d_out, int out_size, void* d_ws, size_t ws_size,
                              hipStream_t stream) {
  const float* Q    = (const float*)d_in[0];
  const float* K    = (const float*)d_in[1];
  const float* V    = (const float*)d_in[2];
  const float* mask = (const float*)d_in[3];

  float* part = (float*)d_ws;                      // BH*CH*4096 floats (~12.6 MB)
  float* ktv  = part + (size_t)BH * CH * 4096;     // BH*4096 floats

  cosattn_p1 <<<BH * CH, 256, 0, stream>>>(K, V, mask, part);
  cosattn_p1r<<<BH,      256, 0, stream>>>(part, ktv);
  cosattn_p2 <<<dim3(Nc / 128, BH), 256, 0, stream>>>(Q, ktv, (float*)d_out);
}
